// PreferencePredictor_52149492908439
// MI455X (gfx1250) — matrix operand order repaired and checked by reading, not compiled, not run on hardware
//
#include <hip/hip_runtime.h>

typedef _Float16 h16 __attribute__((ext_vector_type(16)));
typedef _Float16 h8  __attribute__((ext_vector_type(8)));
typedef float    f8  __attribute__((ext_vector_type(8)));
typedef float    f4  __attribute__((ext_vector_type(4)));

#define DDIM 768
#define HDIM 256

// ---------------------------------------------------------------------------
// Prep 1: convert weights fp32 -> f16 into workspace
// ---------------------------------------------------------------------------
__global__ __launch_bounds__(256)
void pp_convert(const float* __restrict__ Wl,
                const float* __restrict__ inpw,
                const float* __restrict__ opw,
                _Float16* __restrict__ WlH,
                _Float16* __restrict__ WqH,
                _Float16* __restrict__ OpH)
{
    int i = blockIdx.x * 256 + threadIdx.x;
    if (i < HDIM * DDIM) WlH[i] = (_Float16)Wl[i];
    if (i < HDIM * HDIM) {
        WqH[i] = (_Float16)inpw[i];       // rows 0..255 of in_proj_w = Wq_a
        OpH[i] = (_Float16)opw[i];
    }
}

// ---------------------------------------------------------------------------
// Prep 2: u, qv projections and attention K (pre-scaled by 1/sqrt(64)) / V
// ---------------------------------------------------------------------------
__global__ __launch_bounds__(256)
void pp_ctx(const float* __restrict__ user, const float* __restrict__ query,
            const float* __restrict__ Wu, const float* __restrict__ bu,
            const float* __restrict__ Wq, const float* __restrict__ bq,
            const float* __restrict__ inpw, const float* __restrict__ inpb,
            float* __restrict__ kbuf, float* __restrict__ vbuf)
{
    __shared__ float su[HDIM], sq[HDIM];
    const int t = threadIdx.x;
    float au = bu[t], aq = bq[t];
    for (int j = 0; j < DDIM; ++j) {
        au = fmaf(user[j],  Wu[(size_t)t * DDIM + j], au);
        aq = fmaf(query[j], Wq[(size_t)t * DDIM + j], aq);
    }
    su[t] = au; sq[t] = aq;
    __syncthreads();
    const float* wk = inpw + (size_t)(HDIM + t) * HDIM;       // Wk_a row t
    const float* wv = inpw + (size_t)(2 * HDIM + t) * HDIM;   // Wv_a row t
    float dk0 = 0.f, dk1 = 0.f, dv0 = 0.f, dv1 = 0.f;
    for (int j = 0; j < HDIM; ++j) {
        float wkj = wk[j], wvj = wv[j];
        dk0 = fmaf(su[j], wkj, dk0);
        dk1 = fmaf(sq[j], wkj, dk1);
        dv0 = fmaf(su[j], wvj, dv0);
        dv1 = fmaf(sq[j], wvj, dv1);
    }
    const float bk = inpb[HDIM + t], bv = inpb[2 * HDIM + t];
    const float scale = 0.125f;                // 1/sqrt(head_dim=64)
    kbuf[t]        = (dk0 + bk) * scale;       // kk = 0 (u context row)
    kbuf[HDIM + t] = (dk1 + bk) * scale;       // kk = 1 (query context row)
    vbuf[t]        = dv0 + bv;
    vbuf[HDIM + t] = dv1 + bv;
}

// ---------------------------------------------------------------------------
// Main fused kernel: 4 waves/block, 16 llm rows per wave, 64 rows per block
// ---------------------------------------------------------------------------
__global__ __launch_bounds__(128)
void pp_main(const float* __restrict__ llm,
             const float* __restrict__ bl,
             const float* __restrict__ inpb,
             const float* __restrict__ opb,
             const float* __restrict__ Wo,
             const float* __restrict__ bo,
             const _Float16* __restrict__ WlH,
             const _Float16* __restrict__ WqH,
             const _Float16* __restrict__ OpH,
             const float* __restrict__ kbuf,
             const float* __restrict__ vbuf,
             float* __restrict__ out)
{
    __shared__ _Float16 Lh[4][16][HDIM];   // l strip, f16 (32 KB)
    __shared__ _Float16 Qh[4][16][HDIM];   // q -> attended -> scratch (32 KB)

    const int tid  = threadIdx.x;
    const int lane = tid & 31;
    const int w    = tid >> 5;
    const int g    = lane >> 4;            // half-wave group (0/1)
    const int c15  = lane & 15;
    const int mrow0 = blockIdx.x * 64 + w * 16;

    float ps[8];
#pragma unroll
    for (int j = 0; j < 8; ++j) ps[j] = 0.0f;

    // ================= GEMM1: L = llm_tile * Wl^T + bl (K = 768) ==========
    const float* arowbase = llm + (size_t)(mrow0 + c15) * DDIM + g * 8;
#pragma unroll 1
    for (int p = 0; p < 2; ++p) {
        f8 acc[8];
#pragma unroll
        for (int ct = 0; ct < 8; ++ct)
#pragma unroll
            for (int j = 0; j < 8; ++j) acc[ct][j] = 0.0f;

        const int cb = p * 128;
#pragma unroll 1
        for (int kb = 0; kb < 24; ++kb) {
            const float* ap = arowbase + kb * 32;
            __builtin_prefetch(ap + 32, 0, 1);     // global_prefetch_b8 next K-block
            // A frag: ISA 16-bit 16x32 layout (g==0: K 0-7,16-23 ; g==1: K 8-15,24-31)
            f4 a0 = *(const f4*)(ap + 0);
            f4 a1 = *(const f4*)(ap + 4);
            f4 a2 = *(const f4*)(ap + 16);
            f4 a3 = *(const f4*)(ap + 20);
            h16 A;
#pragma unroll
            for (int i = 0; i < 4; ++i) {
                A[i]      = (_Float16)a0[i];
                A[i + 4]  = (_Float16)a1[i];
                A[i + 8]  = (_Float16)a2[i];
                A[i + 12] = (_Float16)a3[i];
            }
#pragma unroll
            for (int ct = 0; ct < 8; ++ct) {
                // B frag: lane = output col n, same K order as A (g==0: K 0-7,16-23 ; g==1: K 8-15,24-31)
                const _Float16* bp = WlH + (size_t)(cb + ct * 16 + c15) * DDIM + kb * 32 + g * 8;
                h8 b0 = *(const h8*)(bp);
                h8 b1 = *(const h8*)(bp + 16);
                h16 B;
#pragma unroll
                for (int i = 0; i < 8; ++i) { B[i] = b0[i]; B[i + 8] = b1[i]; }
                acc[ct] = __builtin_amdgcn_wmma_f32_16x16x32_f16(
                    false, A, false, B, (short)0, acc[ct], false, false);
            }
        }
        // C layout: col = ct*16 + c15, rows g*8 + j  -> bias + store f16 to LDS
#pragma unroll
        for (int ct = 0; ct < 8; ++ct) {
            const int col = cb + ct * 16 + c15;
            const float b = bl[col];
#pragma unroll
            for (int j = 0; j < 8; ++j)
                Lh[w][g * 8 + j][col] = (_Float16)(acc[ct][j] + b);
        }
    }
    __syncthreads();

    // ================= GEMM2: q = L * Wq_a^T + bq_a (K = 256) =============
#pragma unroll 1
    for (int p = 0; p < 2; ++p) {
        f8 acc[8];
#pragma unroll
        for (int ct = 0; ct < 8; ++ct)
#pragma unroll
            for (int j = 0; j < 8; ++j) acc[ct][j] = 0.0f;

        const int cb = p * 128;
#pragma unroll 1
        for (int kb = 0; kb < 8; ++kb) {
            const _Float16* ap = &Lh[w][c15][kb * 32 + g * 8];
            h8 al = *(const h8*)(ap);
            h8 ah = *(const h8*)(ap + 16);
            h16 A;
#pragma unroll
            for (int i = 0; i < 8; ++i) { A[i] = al[i]; A[i + 8] = ah[i]; }
#pragma unroll
            for (int ct = 0; ct < 8; ++ct) {
                const _Float16* bp = WqH + (size_t)(cb + ct * 16 + c15) * HDIM + kb * 32 + g * 8;
                h8 b0 = *(const h8*)(bp);
                h8 b1 = *(const h8*)(bp + 16);
                h16 B;
#pragma unroll
                for (int i = 0; i < 8; ++i) { B[i] = b0[i]; B[i + 8] = b1[i]; }
                acc[ct] = __builtin_amdgcn_wmma_f32_16x16x32_f16(
                    false, A, false, B, (short)0, acc[ct], false, false);
            }
        }
#pragma unroll
        for (int ct = 0; ct < 8; ++ct) {
            const int col = cb + ct * 16 + c15;
            const float b = inpb[col];                 // bq_a
#pragma unroll
            for (int j = 0; j < 8; ++j)
                Qh[w][g * 8 + j][col] = (_Float16)(acc[ct][j] + b);
        }
    }
    __syncthreads();

    // ================= Attention: softmax over 2 context keys =============
    // lane owns (row = c15, heads {2g, 2g+1}); overwrites q with attended in-place
#pragma unroll
    for (int hh = 0; hh < 2; ++hh) {
        const int h = g * 2 + hh;
        h8 qr[8];
#pragma unroll
        for (int c = 0; c < 8; ++c)
            qr[c] = *(const h8*)&Qh[w][c15][h * 64 + c * 8];
        const float* k0 = kbuf + h * 64;               // pre-scaled
        const float* k1 = kbuf + HDIM + h * 64;
        float s0 = 0.f, s1 = 0.f;
#pragma unroll
        for (int c = 0; c < 8; ++c)
#pragma unroll
            for (int e = 0; e < 8; ++e) {
                const float qv = (float)qr[c][e];
                const int d = c * 8 + e;
                s0 = fmaf(qv, k0[d], s0);
                s1 = fmaf(qv, k1[d], s1);
            }
        const float mx = fmaxf(s0, s1);
        const float e0 = __expf(s0 - mx), e1 = __expf(s1 - mx);
        const float inv = 1.0f / (e0 + e1);
        const float a0 = e0 * inv, a1 = e1 * inv;
        const float* v0 = vbuf + h * 64;
        const float* v1 = vbuf + HDIM + h * 64;
#pragma unroll
        for (int c = 0; c < 8; ++c) {
            h8 o;
#pragma unroll
            for (int e = 0; e < 8; ++e) {
                const int d = c * 8 + e;
                o[e] = (_Float16)(fmaf(a0, v0[d], a1 * v1[d]));
            }
            *(h8*)&Qh[w][c15][h * 64 + c * 8] = o;
        }
    }
    __syncthreads();

    // ====== GEMM3: attended * out_proj^T + opb, fused relu·Wo reduction ===
#pragma unroll 1
    for (int p = 0; p < 2; ++p) {
        f8 acc[8];
#pragma unroll
        for (int ct = 0; ct < 8; ++ct)
#pragma unroll
            for (int j = 0; j < 8; ++j) acc[ct][j] = 0.0f;

        const int cb = p * 128;
#pragma unroll 1
        for (int kb = 0; kb < 8; ++kb) {
            const _Float16* ap = &Qh[w][c15][kb * 32 + g * 8];
            h8 al = *(const h8*)(ap);
            h8 ah = *(const h8*)(ap + 16);
            h16 A;
#pragma unroll
            for (int i = 0; i < 8; ++i) { A[i] = al[i]; A[i + 8] = ah[i]; }
#pragma unroll
            for (int ct = 0; ct < 8; ++ct) {
                const _Float16* bp = OpH + (size_t)(cb + ct * 16 + c15) * HDIM + kb * 32 + g * 8;
                h8 b0 = *(const h8*)(bp);
                h8 b1 = *(const h8*)(bp + 16);
                h16 B;
#pragma unroll
                for (int i = 0; i < 8; ++i) { B[i] = b0[i]; B[i + 8] = b1[i]; }
                acc[ct] = __builtin_amdgcn_wmma_f32_16x16x32_f16(
                    false, A, false, B, (short)0, acc[ct], false, false);
            }
        }
        // fold relu(attended_out)*Wo[col] + relu(l)*Wo[256+col] into row partials
#pragma unroll
        for (int ct = 0; ct < 8; ++ct) {
            const int col = cb + ct * 16 + c15;
            const float ob  = opb[col];
            const float wo1 = Wo[col];
            const float wo2 = Wo[HDIM + col];
#pragma unroll
            for (int j = 0; j < 8; ++j) {
                const float av = acc[ct][j] + ob;
                const float lv = (float)Lh[w][g * 8 + j][col];
                ps[j] = fmaf(fmaxf(av, 0.f), wo1,
                        fmaf(fmaxf(lv, 0.f), wo2, ps[j]));
            }
        }
    }
    __syncthreads();

    // cross-lane reduce via scratch overlaid on Qh (free now)
    float* scratch = (float*)&Qh[w][0][0];   // 16 rows x 16 partials (1 KB)
#pragma unroll
    for (int j = 0; j < 8; ++j)
        scratch[(g * 8 + j) * 16 + c15] = ps[j];
    __syncthreads();

    if (lane < 16) {
        float s = bo[0];
#pragma unroll
        for (int c = 0; c < 16; ++c) s += scratch[lane * 16 + c];
        out[mrow0 + lane] = s;
    }
}

// ---------------------------------------------------------------------------
extern "C" void kernel_launch(void* const* d_in, const int* in_sizes, int n_in,
                              void* d_out, int out_size, void* d_ws, size_t ws_size,
                              hipStream_t stream)
{
    const float* user  = (const float*)d_in[0];
    const float* query = (const float*)d_in[1];
    const float* llm   = (const float*)d_in[2];
    const float* Wu    = (const float*)d_in[3];
    const float* bu    = (const float*)d_in[4];
    const float* Wq    = (const float*)d_in[5];
    const float* bq    = (const float*)d_in[6];
    const float* Wl    = (const float*)d_in[7];
    const float* bl    = (const float*)d_in[8];
    const float* inpw  = (const float*)d_in[9];
    const float* inpb  = (const float*)d_in[10];
    const float* opw   = (const float*)d_in[11];
    const float* opb   = (const float*)d_in[12];
    const float* Wo    = (const float*)d_in[13];
    const float* bo    = (const float*)d_in[14];
    float* out = (float*)d_out;

    char* ws = (char*)d_ws;
    _Float16* WlH = (_Float16*)(ws);                  // 256*768*2 = 393216 B
    _Float16* WqH = (_Float16*)(ws + 393216);         // 256*256*2 = 131072 B
    _Float16* OpH = (_Float16*)(ws + 524288);         // 131072 B
    float*    kbf = (float*)   (ws + 655360);         // 2*256*4 = 2048 B
    float*    vbf = (float*)   (ws + 657408);         // 2048 B

    pp_convert<<<(HDIM * DDIM + 255) / 256, 256, 0, stream>>>(Wl, inpw, opw, WlH, WqH, OpH);
    pp_ctx<<<1, 256, 0, stream>>>(user, query, Wu, bu, Wq, bq, inpw, inpb, kbf, vbf);
    const int m = in_sizes[2] / DDIM;                 // 131072
    pp_main<<<m / 64, 128, 0, stream>>>(llm, bl, inpb, opb, Wo, bo,
                                        WlH, WqH, OpH, kbf, vbf, out);
}
